// model_10204842295856
// MI455X (gfx1250) — matrix operand order repaired and checked by reading, not compiled, not run on hardware
//
#include <hip/hip_runtime.h>
#include <hip/hip_bf16.h>
#include <math.h>

typedef _Float16 v16h __attribute__((ext_vector_type(16)));
typedef _Float16 v8h  __attribute__((ext_vector_type(8)));
typedef float    v8f  __attribute__((ext_vector_type(8)));

#define DD      32      // state dim
#define PP      8       // obs dim
#define MSTEPS  50
#define HH      128     // hidden
#define NN      32768
#define KF      64      // feature K padded 42 -> 64

// ---- gfx1250 transcendental tanh (guarded) ----
__device__ __forceinline__ float fast_tanhf(float x) {
#if __has_builtin(__builtin_amdgcn_tanhf)
  return __builtin_amdgcn_tanhf(x);
#elif __has_builtin(__builtin_amdgcn_tanh_f32)
  return __builtin_amdgcn_tanh_f32(x);
#else
  return tanhf(x);
#endif
}

// sum across the 16 lanes of each half-wave via ds_swizzle xor 1,2,4,8
__device__ __forceinline__ float reduce16(float x) {
  x += __int_as_float(__builtin_amdgcn_ds_swizzle(__float_as_int(x), 0x041F));
  x += __int_as_float(__builtin_amdgcn_ds_swizzle(__float_as_int(x), 0x081F));
  x += __int_as_float(__builtin_amdgcn_ds_swizzle(__float_as_int(x), 0x101F));
  x += __int_as_float(__builtin_amdgcn_ds_swizzle(__float_as_int(x), 0x201F));
  return x;
}

__device__ __forceinline__ v8f wmma_f16(v16h a, v16h b, v8f c) {
  // D = A(16x32 f16) * B(32x16 f16) + C(16x16 f32)
  return __builtin_amdgcn_wmma_f32_16x16x32_f16(false, a, false, b, (short)0, c,
                                                false, false);
}

__device__ __forceinline__ v16h cat8(v8h lo, v8h hi) {
  return __builtin_shufflevector(lo, hi, 0,1,2,3,4,5,6,7,8,9,10,11,12,13,14,15);
}

__global__ __launch_bounds__(256)
void bsde_scan_kernel(const float* __restrict__ obs,     // (P)
                      const float* __restrict__ X0,      // (N,D)
                      const float* __restrict__ V0i,     // (N)
                      const float* __restrict__ noise,   // (M,N,D)
                      const float* __restrict__ W1g,     // (42,H)
                      const float* __restrict__ b1g,     // (H)
                      const float* __restrict__ W2g,     // (H,D)
                      const float* __restrict__ b2g,     // (D)
                      const float* __restrict__ thetap,  // (1)
                      const int*   __restrict__ obsidxp, // (1)
                      const int*   __restrict__ crp,     // (1)
                      float* __restrict__ outX,          // (N,D)
                      float* __restrict__ outV)          // (N)
{
  // ---- LDS: weight fragments (block-shared) + per-wave staging ----
  // B-fragment layout assumed (16x16x32 f16): lane holds N = lane%16,
  // K = 8*(lane/16) + {0..7}, 16 + 8*(lane/16) + {0..7} -> 16 halves/lane.
  __shared__ __align__(32) _Float16 sW1[16][32][16];  // frag = nt*2+kc, 16 KB
  __shared__ __align__(32) _Float16 sW2[8][32][16];   // frag = kc*2+nt,  8 KB
  __shared__ __align__(32) _Float16 sFeat[8][16][KF]; // per-wave 16x64,  16 KB
  __shared__ __align__(32) _Float16 sHc[8][16][32];   // per-wave h chunk, 8 KB

  const int tid  = threadIdx.x;
  const int lane = tid & 31;
  const int wv   = tid >> 5;
  const int half = lane >> 4;
  const int lq   = lane & 15;
  const int mrow = 8 * half;                 // C-layout row offset for this half
  const int pbase = (blockIdx.x * 8 + wv) * 16;

  const float theta = thetap[0];
  const float cr    = (float)crp[0];
  const float tf    = (float)obsidxp[0];
  const float dt    = 1.0f / (float)MSTEPS;  // INTERVAL / M
  const float sdt   = sqrtf(dt);

  // ---- stage W1 into B-fragment layout (zero-pad K=42..63) ----
  #pragma unroll 2
  for (int i = tid; i < 16 * 32 * 16; i += 256) {
    int f  = i >> 9;
    int ln = (i >> 4) & 31;
    int j  = i & 15;
    int nt = f >> 1, kc = f & 1;
    int K  = kc * 32 + 8 * (ln >> 4) + (j & 7) + 16 * (j >> 3);
    int Nn = nt * 16 + (ln & 15);
    float w = (K < DD + PP + 2) ? W1g[K * HH + Nn] : 0.0f;
    sW1[f][ln][j] = (_Float16)w;
  }
  // ---- stage W2 ----
  #pragma unroll 2
  for (int i = tid; i < 8 * 32 * 16; i += 256) {
    int f  = i >> 9;
    int ln = (i >> 4) & 31;
    int j  = i & 15;
    int kc = f >> 1, nt = f & 1;
    int K  = kc * 32 + 8 * (ln >> 4) + (j & 7) + 16 * (j >> 3);
    int Nn = nt * 16 + (ln & 15);
    sW2[f][ln][j] = (_Float16)W2g[K * DD + Nn];
  }
  // ---- per-wave constant feature columns: Y (32..39), t (41), pad (42..63) ----
  for (int i = lane; i < 16 * 32; i += 32) {
    int r = i >> 5;
    int c = 32 + (i & 31);
    float v = 0.0f;
    if (c < 40)      v = obs[c - 32];
    else if (c == 41) v = tf;
    sFeat[wv][r][c] = (_Float16)v;   // col 40 (s) rewritten every step
  }
  __syncthreads();

  // ---- biases in C-layout lanes ----
  float vb1[8], vb2[2];
  #pragma unroll
  for (int nt = 0; nt < 8; ++nt) vb1[nt] = b1g[nt * 16 + lq];
  vb2[0] = b2g[lq];
  vb2[1] = b2g[16 + lq];

  // ---- load state in WMMA C layout: element(v, lane) = (m = v+8*half, d = dtile*16+lq)
  float X[2][8], vV[8];
  #pragma unroll
  for (int v = 0; v < 8; ++v) {
    int m = pbase + v + mrow;
    X[0][v] = X0[m * DD + lq];
    X[1][v] = X0[m * DD + 16 + lq];
    vV[v]   = V0i[m];
  }

  // ================= 50-step scan =================
  for (int step = 0; step < MSTEPS; ++step) {
    const float s = dt * (float)step;
    const float* np_ = noise + (size_t)step * NN * DD;

    // Brownian increments, loaded directly in C layout
    float wn[2][8];
    #pragma unroll
    for (int v = 0; v < 8; ++v) {
      int m = pbase + v + mrow;
      wn[0][v] = sdt * np_[m * DD + lq];
      wn[1][v] = sdt * np_[m * DD + 16 + lq];
    }
    if (step + 1 < MSTEPS) {  // prefetch next step's 2KB wave region (64B/lane)
      __builtin_prefetch(noise + (size_t)(step + 1) * NN * DD +
                         (size_t)pbase * DD + lane * 16, 0, 0);
    }

    // write dynamic feature columns: X (0..31) and s (40)
    #pragma unroll
    for (int v = 0; v < 8; ++v) {
      sFeat[wv][v + mrow][lq]      = (_Float16)X[0][v];
      sFeat[wv][v + mrow][16 + lq] = (_Float16)X[1][v];
    }
    sFeat[wv][lq][40] = (_Float16)s;
    __builtin_amdgcn_wave_barrier();

    // A1 fragments (16x32 f16 A layout: lane holds row lq; halves hold
    // K = kc*32 + {8*half..8*half+7} and {16+8*half..23+8*half})
    v16h a1[2];
    #pragma unroll
    for (int kc = 0; kc < 2; ++kc) {
      v8h lo = *(const v8h*)&sFeat[wv][lq][kc * 32 + 8 * half];
      v8h hi = *(const v8h*)&sFeat[wv][lq][kc * 32 + 16 + 8 * half];
      a1[kc] = cat8(lo, hi);
    }

    // Z accumulators pre-seeded with b2 (bias per column = per lane)
    v8f accZ[2];
    #pragma unroll
    for (int nt = 0; nt < 2; ++nt)
      #pragma unroll
      for (int k = 0; k < 8; ++k) accZ[nt][k] = vb2[nt];

    // fused MLP: for each 32-wide hidden chunk c, do GEMM1 tiles 2c,2c+1,
    // tanh-stage to LDS, then GEMM2 k-chunk c into Z
    #pragma unroll
    for (int c = 0; c < 4; ++c) {
      #pragma unroll
      for (int t2 = 0; t2 < 2; ++t2) {
        int nt = 2 * c + t2;
        v16h bA = *(const v16h*)&sW1[nt * 2 + 0][lane][0];
        v16h bB = *(const v16h*)&sW1[nt * 2 + 1][lane][0];
        v8f h = {};
        h = wmma_f16(a1[0], bA, h);
        h = wmma_f16(a1[1], bB, h);
        #pragma unroll
        for (int k = 0; k < 8; ++k) {
          float hv = fast_tanhf(h[k] + vb1[nt]);
          sHc[wv][k + mrow][t2 * 16 + lq] = (_Float16)hv;  // C->row-major f16
        }
      }
      __builtin_amdgcn_wave_barrier();
      // A2 fragment from staged h chunk
      v8h lo = *(const v8h*)&sHc[wv][lq][8 * half];
      v8h hi = *(const v8h*)&sHc[wv][lq][16 + 8 * half];
      v16h a2 = cat8(lo, hi);
      #pragma unroll
      for (int nt = 0; nt < 2; ++nt) {
        v16h bw = *(const v16h*)&sW2[c * 2 + nt][lane][0];
        accZ[nt] = wmma_f16(a2, bw, accZ[nt]);
      }
      __builtin_amdgcn_wave_barrier();  // LDS in-order per wave: WAR on sHc safe
    }

    // V update: drift_V = (0.5 - cr) * sum(Z^2); plus sum(Z * W)
    // X update: X += dt*(-theta*X - cr*Z) + W     (SIGMA = 1)
    #pragma unroll
    for (int v = 0; v < 8; ++v) {
      float z0 = accZ[0][v], z1 = accZ[1][v];
      float q = z0 * z0 + z1 * z1;
      float r = z0 * wn[0][v] + z1 * wn[1][v];
      q = reduce16(q);   // per-particle sum over d (d is along lanes)
      r = reduce16(r);
      vV[v] += dt * (0.5f - cr) * q + r;
      X[0][v] = X[0][v] + dt * (-theta * X[0][v] - cr * z0) + wn[0][v];
      X[1][v] = X[1][v] + dt * (-theta * X[1][v] - cr * z1) + wn[1][v];
    }
  }

  // ---- write results ----
  #pragma unroll
  for (int v = 0; v < 8; ++v) {
    int m = pbase + v + mrow;
    outX[m * DD + lq]      = X[0][v];
    outX[m * DD + 16 + lq] = X[1][v];
    if (lq == 0) outV[m] = vV[v];   // lanes 0 and 16 carry the reduced sums
  }
}

extern "C" void kernel_launch(void* const* d_in, const int* in_sizes, int n_in,
                              void* d_out, int out_size, void* d_ws, size_t ws_size,
                              hipStream_t stream) {
  const float* obs    = (const float*)d_in[0];
  const float* X0     = (const float*)d_in[1];
  const float* V0i    = (const float*)d_in[2];
  const float* noise  = (const float*)d_in[3];
  const float* W1     = (const float*)d_in[4];
  const float* b1     = (const float*)d_in[5];
  const float* W2     = (const float*)d_in[6];
  const float* b2     = (const float*)d_in[7];
  const float* theta  = (const float*)d_in[8];
  const int*   obsidx = (const int*)d_in[9];
  const int*   crq    = (const int*)d_in[10];

  float* outX = (float*)d_out;            // (N, D) first
  float* outV = outX + (size_t)NN * DD;   // then (N,)

  dim3 grid(NN / 128);   // 8 waves/block * 16 particles/wave = 128 particles/block
  dim3 block(256);
  bsde_scan_kernel<<<grid, block, 0, stream>>>(obs, X0, V0i, noise, W1, b1, W2, b2,
                                               theta, obsidx, crq, outX, outV);
}
